// SAGEConvLayer_86380382257211
// MI455X (gfx1250) — matrix operand order repaired and checked by reading, not compiled, not run on hardware
//
#include <hip/hip_runtime.h>

#define NP 100000
#define NA 50000
#define D  128
#define NE 1600000
#define BN_EPS 1e-5f

typedef __attribute__((ext_vector_type(16))) _Float16 v16h;
typedef __attribute__((ext_vector_type(8)))  float    v8f;

// Packed weight matrix: per (ntile 0..7, kchunk 0..3, lane 0..31) one v16h
// fragment, laid out contiguously -> one 32B vector load per B fragment.
#define PACKW_ELEMS (8 * 4 * 32 * 16)   // = 16384 halfs per 128x128 matrix

// ---------------------------------------------------------------------------
// WMMA helper: D = A(16x32 f16) * B(32x16 f16) + C(16x16 f32)
// ---------------------------------------------------------------------------
__device__ __forceinline__ v8f wmma_f16(v16h a, v16h b, v8f c) {
    return __builtin_amdgcn_wmma_f32_16x16x32_f16(
        /*neg_a=*/false, a, /*neg_b=*/false, b,
        /*c_mod=*/(short)0, c, /*reuse_a=*/false, /*reuse_b=*/false);
}

// A fragment from an LDS-staged 16x128 f16 tile.
// Lane L (lm = L&15 row, hi = L>>4): halfs 0..7  = K kc*32+hi*8 .. +7 (contig)
//                                    halfs 8..15 = K kc*32+16+hi*8 .. +7 (contig)
__device__ __forceinline__ v16h lds_a_frag(const _Float16* __restrict__ sA,
                                           int lm, int kc, int hi) {
    int base = lm * D + kc * 32 + hi * 8;
    union { float4 f4[2]; v16h v; } u;
    u.f4[0] = *(const float4*)(sA + base);
    u.f4[1] = *(const float4*)(sA + base + 16);
    return u.v;
}

// B fragment: single 32-byte vector load from pre-packed weights.
__device__ __forceinline__ v16h packed_b_frag(const _Float16* __restrict__ pW,
                                              int ntile, int kc, int lane) {
    const v16h* p = (const v16h*)pW;
    return p[(ntile * 4 + kc) * 32 + lane];
}

// Cooperative stage: 8 consecutive floats -> 8 f16 in LDS (16B aligned store).
__device__ __forceinline__ void stage8(const float* __restrict__ src, float scale,
                                       _Float16* __restrict__ dst) {
    float4 f0 = *(const float4*)(src);
    float4 f1 = *(const float4*)(src + 4);
    union { _Float16 h[8]; float4 f; } u;
    u.h[0] = (_Float16)(f0.x * scale); u.h[1] = (_Float16)(f0.y * scale);
    u.h[2] = (_Float16)(f0.z * scale); u.h[3] = (_Float16)(f0.w * scale);
    u.h[4] = (_Float16)(f1.x * scale); u.h[5] = (_Float16)(f1.y * scale);
    u.h[6] = (_Float16)(f1.z * scale); u.h[7] = (_Float16)(f1.w * scale);
    *(float4*)dst = u.f;
}

__device__ __forceinline__ float fast_inv_cnt(float c) {
    return __builtin_amdgcn_rcpf(fmaxf(c, 1.0f));   // v_rcp_f32, exact enough
}

// ---------------------------------------------------------------------------
// Pack a 128x128 f32 weight (row-major [K][N]) into WMMA B-fragment order f16.
// Fragment half j of lane L in (ntile, kchunk): k = kc*32 + (j>>3)*16 + (L>>4)*8 + (j&7),
//                                               n = ntile*16 + (L&15).
// ---------------------------------------------------------------------------
__global__ void pack_w_kernel(const float* __restrict__ W, _Float16* __restrict__ out) {
    int t = blockIdx.x * blockDim.x + threadIdx.x;
    if (t >= PACKW_ELEMS) return;
    int j    = t & 15;
    int lane = (t >> 4) & 31;
    int kc   = (t >> 9) & 3;
    int nt   = t >> 11;
    int k = kc * 32 + (j >> 3) * 16 + (lane >> 4) * 8 + (j & 7);
    int n = nt * 16 + (lane & 15);
    out[t] = (_Float16)W[k * D + n];
}

// ---------------------------------------------------------------------------
// Zero workspace
// ---------------------------------------------------------------------------
__global__ void zero_kernel(float* p, long n) {
    long stride = (long)gridDim.x * blockDim.x;
    for (long i = (long)blockIdx.x * blockDim.x + threadIdx.x; i < n; i += stride)
        p[i] = 0.0f;
}

// ---------------------------------------------------------------------------
// Edge scatter: one wave per edge, lane l handles float4 chunk l (4 cols).
// ---------------------------------------------------------------------------
__global__ void scatter_sum_kernel(const float4* __restrict__ xsrc4,
                                   const int* __restrict__ src,
                                   const int* __restrict__ dst,
                                   float* __restrict__ sum,
                                   float* __restrict__ cnt,
                                   int nE) {
    long total  = (long)nE * 32;
    long stride = (long)gridDim.x * blockDim.x;
    for (long idx = (long)blockIdx.x * blockDim.x + threadIdx.x; idx < total; idx += stride) {
        int e = (int)(idx >> 5);
        int l = (int)(idx & 31);
        int s = src[e];
        int d = dst[e];
        float4 v = xsrc4[(long)s * 32 + l];
        float* p = sum + (long)d * D + l * 4;
        atomicAdd(p + 0, v.x);
        atomicAdd(p + 1, v.y);
        atomicAdd(p + 2, v.z);
        atomicAdd(p + 3, v.w);
        if (l == 0) atomicAdd(cnt + d, 1.0f);
    }
}

// ---------------------------------------------------------------------------
// Paper GEMM: h = relu( mean_c@Wl_c + bl_c + x@Wr_c + mean_w@Wl_w + bl_w + x@Wr_w )
// Block = 256 threads = 8 waves; block -> 16 rows, wave w -> 16-col tile w.
// A tiles staged in LDS (f16, pre-scaled) and shared by all 8 waves.
// ---------------------------------------------------------------------------
__global__ void __launch_bounds__(256)
sage_gemm_paper(const float* __restrict__ x_paper,
                const float* __restrict__ sum_c, const float* __restrict__ cnt_c,
                const _Float16* __restrict__ pWl_c, const float* __restrict__ bl_c,
                const _Float16* __restrict__ pWr_c,
                const float* __restrict__ sum_w, const float* __restrict__ cnt_w,
                const _Float16* __restrict__ pWl_w, const float* __restrict__ bl_w,
                const _Float16* __restrict__ pWr_w,
                float* __restrict__ out) {
    __shared__ __align__(16) _Float16 sMC[16 * D];   // mean_cites tile
    __shared__ __align__(16) _Float16 sMW[16 * D];   // mean_writes tile
    __shared__ __align__(16) _Float16 sX [16 * D];   // x_paper tile

    int tid  = threadIdx.x;
    int m0   = blockIdx.x * 16;

    // Stage: thread t handles row t>>4, cols (t&15)*8 .. +7
    {
        int row  = tid >> 4;
        int col0 = (tid & 15) * 8;
        int m    = m0 + row;
        long off = (long)m * D + col0;
        float ic = fast_inv_cnt(cnt_c[m]);
        float iw = fast_inv_cnt(cnt_w[m]);
        stage8(sum_c  + off, ic,   sMC + row * D + col0);
        stage8(sum_w  + off, iw,   sMW + row * D + col0);
        stage8(x_paper + off, 1.0f, sX + row * D + col0);
    }
    __syncthreads();

    int lane = tid & 31;
    int wv   = tid >> 5;             // 0..7 -> column tile
    int lm   = lane & 15;
    int hi   = lane >> 4;
    int n    = wv * 16 + lm;

    v8f c;
    float binit = bl_c[n] + bl_w[n];
#pragma unroll
    for (int r = 0; r < 8; ++r) c[r] = binit;

#pragma unroll
    for (int kc = 0; kc < 4; ++kc) {
        v16h a_mc = lds_a_frag(sMC, lm, kc, hi);
        c = wmma_f16(a_mc, packed_b_frag(pWl_c, wv, kc, lane), c);

        v16h a_mw = lds_a_frag(sMW, lm, kc, hi);
        c = wmma_f16(a_mw, packed_b_frag(pWl_w, wv, kc, lane), c);

        v16h a_x = lds_a_frag(sX, lm, kc, hi);
        c = wmma_f16(a_x, packed_b_frag(pWr_c, wv, kc, lane), c);
        c = wmma_f16(a_x, packed_b_frag(pWr_w, wv, kc, lane), c);
    }

#pragma unroll
    for (int r = 0; r < 8; ++r) {
        int row = m0 + r + hi * 8;
        out[(long)row * D + n] = fmaxf(c[r], 0.0f);   // fused ReLU
    }
}

// ---------------------------------------------------------------------------
// Author GEMM: h = relu( mean_rev@Wl + bl + x_author@Wr )
// ---------------------------------------------------------------------------
__global__ void __launch_bounds__(256)
sage_gemm_author(const float* __restrict__ x_author,
                 const float* __restrict__ sum_r, const float* __restrict__ cnt_r,
                 const _Float16* __restrict__ pWl_r, const float* __restrict__ bl_r,
                 const _Float16* __restrict__ pWr_r,
                 float* __restrict__ out) {
    __shared__ __align__(16) _Float16 sM[16 * D];
    __shared__ __align__(16) _Float16 sX[16 * D];

    int tid = threadIdx.x;
    int m0  = blockIdx.x * 16;

    {
        int row  = tid >> 4;
        int col0 = (tid & 15) * 8;
        int m    = m0 + row;
        long off = (long)m * D + col0;
        float ir = fast_inv_cnt(cnt_r[m]);
        stage8(sum_r   + off, ir,   sM + row * D + col0);
        stage8(x_author + off, 1.0f, sX + row * D + col0);
    }
    __syncthreads();

    int lane = tid & 31;
    int wv   = tid >> 5;
    int lm   = lane & 15;
    int hi   = lane >> 4;
    int n    = wv * 16 + lm;

    v8f c;
    float binit = bl_r[n];
#pragma unroll
    for (int r = 0; r < 8; ++r) c[r] = binit;

#pragma unroll
    for (int kc = 0; kc < 4; ++kc) {
        v16h a_m = lds_a_frag(sM, lm, kc, hi);
        c = wmma_f16(a_m, packed_b_frag(pWl_r, wv, kc, lane), c);

        v16h a_x = lds_a_frag(sX, lm, kc, hi);
        c = wmma_f16(a_x, packed_b_frag(pWr_r, wv, kc, lane), c);
    }

#pragma unroll
    for (int r = 0; r < 8; ++r) {
        int row = m0 + r + hi * 8;
        out[(long)row * D + n] = fmaxf(c[r], 0.0f);
    }
}

// ---------------------------------------------------------------------------
// BatchNorm column stats
// ---------------------------------------------------------------------------
__global__ void bn_stats_kernel(const float* __restrict__ h, long nNodes,
                                float* __restrict__ colsum, float* __restrict__ colsq) {
    __shared__ float ssum[D];
    __shared__ float ssq[D];
    if (threadIdx.x < D) { ssum[threadIdx.x] = 0.0f; ssq[threadIdx.x] = 0.0f; }
    __syncthreads();

    long total  = nNodes * D;
    long stride = (long)gridDim.x * blockDim.x;
    for (long idx = (long)blockIdx.x * blockDim.x + threadIdx.x; idx < total; idx += stride) {
        float v = h[idx];
        int c = (int)(idx & (D - 1));
        atomicAdd(&ssum[c], v);
        atomicAdd(&ssq[c], v * v);
    }
    __syncthreads();
    if (threadIdx.x < D) {
        atomicAdd(&colsum[threadIdx.x], ssum[threadIdx.x]);
        atomicAdd(&colsq[threadIdx.x],  ssq[threadIdx.x]);
    }
}

// ---------------------------------------------------------------------------
// BatchNorm apply (in place)
// ---------------------------------------------------------------------------
__global__ void bn_apply_kernel(float* __restrict__ h, long nNodes,
                                const float* __restrict__ colsum,
                                const float* __restrict__ colsq,
                                const float* __restrict__ g,
                                const float* __restrict__ b) {
    float invN = 1.0f / (float)nNodes;
    long total  = nNodes * D;
    long stride = (long)gridDim.x * blockDim.x;
    for (long idx = (long)blockIdx.x * blockDim.x + threadIdx.x; idx < total; idx += stride) {
        int c = (int)(idx & (D - 1));
        float mu  = colsum[c] * invN;
        float var = colsq[c] * invN - mu * mu;
        h[idx] = (h[idx] - mu) * rsqrtf(var + BN_EPS) * g[c] + b[c];
    }
}

// ---------------------------------------------------------------------------
// Host launcher
// ---------------------------------------------------------------------------
extern "C" void kernel_launch(void* const* d_in, const int* in_sizes, int n_in,
                              void* d_out, int out_size, void* d_ws, size_t ws_size,
                              hipStream_t stream) {
    // setup_inputs() dict order
    const float* x_paper   = (const float*)d_in[0];
    const float* x_author  = (const float*)d_in[1];
    const int*   ei_c_src  = (const int*)d_in[2];
    const int*   ei_c_dst  = (const int*)d_in[3];
    const int*   ei_w_src  = (const int*)d_in[4];
    const int*   ei_w_dst  = (const int*)d_in[5];
    const int*   ei_r_src  = (const int*)d_in[6];
    const int*   ei_r_dst  = (const int*)d_in[7];
    const float* Wl_c      = (const float*)d_in[8];
    const float* bl_c      = (const float*)d_in[9];
    const float* Wr_c      = (const float*)d_in[10];
    const float* Wl_w      = (const float*)d_in[11];
    const float* bl_w      = (const float*)d_in[12];
    const float* Wr_w      = (const float*)d_in[13];
    const float* Wl_r      = (const float*)d_in[14];
    const float* bl_r      = (const float*)d_in[15];
    const float* Wr_r      = (const float*)d_in[16];
    const float* bn_g_p    = (const float*)d_in[17];
    const float* bn_b_p    = (const float*)d_in[18];
    const float* bn_g_a    = (const float*)d_in[19];
    const float* bn_b_a    = (const float*)d_in[20];

    // Workspace layout (floats first, then 32B-aligned packed f16 weights)
    float* ws        = (float*)d_ws;
    float* sum_c     = ws;                              // NP*D
    float* sum_w     = sum_c + (long)NP * D;            // NP*D
    float* sum_r     = sum_w + (long)NP * D;            // NA*D
    float* cnt_c     = sum_r + (long)NA * D;            // NP
    float* cnt_w     = cnt_c + NP;                      // NP
    float* cnt_r     = cnt_w + NP;                      // NA
    float* colsum_p  = cnt_r + NA;                      // 128
    float* colsq_p   = colsum_p + D;                    // 128
    float* colsum_a  = colsq_p + D;                     // 128
    float* colsq_a   = colsum_a + D;                    // 128
    long   ws_floats = ((long)NP * D) * 2 + (long)NA * D + 2L * NP + NA + 4 * D;

    uintptr_t pk = (uintptr_t)(colsq_a + D);
    pk = (pk + 31) & ~(uintptr_t)31;
    _Float16* packW = (_Float16*)pk;
    _Float16* pWl_c = packW + 0L * PACKW_ELEMS;
    _Float16* pWr_c = packW + 1L * PACKW_ELEMS;
    _Float16* pWl_w = packW + 2L * PACKW_ELEMS;
    _Float16* pWr_w = packW + 3L * PACKW_ELEMS;
    _Float16* pWl_r = packW + 4L * PACKW_ELEMS;
    _Float16* pWr_r = packW + 5L * PACKW_ELEMS;

    float* h_paper  = (float*)d_out;                    // NP*D
    float* h_author = h_paper + (long)NP * D;           // NA*D

    // 1. Zero accumulators + stats; pack weights to f16 fragment order
    zero_kernel<<<2048, 256, 0, stream>>>(ws, ws_floats);
    {
        int pb = (PACKW_ELEMS + 255) / 256;
        pack_w_kernel<<<pb, 256, 0, stream>>>(Wl_c, pWl_c);
        pack_w_kernel<<<pb, 256, 0, stream>>>(Wr_c, pWr_c);
        pack_w_kernel<<<pb, 256, 0, stream>>>(Wl_w, pWl_w);
        pack_w_kernel<<<pb, 256, 0, stream>>>(Wr_w, pWr_w);
        pack_w_kernel<<<pb, 256, 0, stream>>>(Wl_r, pWl_r);
        pack_w_kernel<<<pb, 256, 0, stream>>>(Wr_r, pWr_r);
    }

    // 2. Edge scatter (bandwidth/atomic bound phase)
    {
        long work = (long)NE * 32;
        int blocks = (int)((work + 255) / 256);
        scatter_sum_kernel<<<blocks, 256, 0, stream>>>(
            (const float4*)x_paper,  ei_c_src, ei_c_dst, sum_c, cnt_c, NE);
        scatter_sum_kernel<<<blocks, 256, 0, stream>>>(
            (const float4*)x_author, ei_w_src, ei_w_dst, sum_w, cnt_w, NE);
        scatter_sum_kernel<<<blocks, 256, 0, stream>>>(
            (const float4*)x_paper,  ei_r_src, ei_r_dst, sum_r, cnt_r, NE);
    }

    // 3. WMMA GEMMs with LDS-staged A tiles, fused bias + edge-type sum + ReLU
    sage_gemm_paper<<<NP / 16, 256, 0, stream>>>(
        x_paper, sum_c, cnt_c, pWl_c, bl_c, pWr_c,
        sum_w, cnt_w, pWl_w, bl_w, pWr_w, h_paper);
    sage_gemm_author<<<NA / 16, 256, 0, stream>>>(
        x_author, sum_r, cnt_r, pWl_r, bl_r, pWr_r, h_author);

    // 4. BatchNorm stats then in-place apply, per node type
    bn_stats_kernel<<<1024, 256, 0, stream>>>(h_paper, (long)NP, colsum_p, colsq_p);
    bn_stats_kernel<<<1024, 256, 0, stream>>>(h_author, (long)NA, colsum_a, colsq_a);
    bn_apply_kernel<<<1024, 256, 0, stream>>>(h_paper, (long)NP, colsum_p, colsq_p, bn_g_p, bn_b_p);
    bn_apply_kernel<<<1024, 256, 0, stream>>>(h_author, (long)NA, colsum_a, colsq_a, bn_g_a, bn_b_a);
}
